// NGCNetGrid_17257178595548
// MI455X (gfx1250) — matrix operand order repaired and checked by reading, not compiled, not run on hardware
//
#include <hip/hip_runtime.h>
#include <hip/hip_bf16.h>
#include <math.h>

typedef __attribute__((ext_vector_type(16))) _Float16 v16h;
typedef __attribute__((ext_vector_type(8)))  float    v8f;

#define TILE 64
#define LDX  136   // f16 row stride for x buffers (272B = 17*16, keeps 16B alignment)
#define LDH  67    // f32 row stride for decoder hidden (odd -> conflict-free column reads)

union AF { v16h v; uint4 q[2]; };
union CF { v8f  v; float f[8]; };

// ---- WMMA fragment loaders (CDNA5 16-bit layouts, CDNA5 ISA 7.12.2) ----
// A 16x32 f16: lanes0-15 row M=lane, K: [0..7]+[16..23]; lanes16-31 row M=lane-16, K: [8..15]+[24..31]
__device__ __forceinline__ v16h load_a_lds(const _Float16* base, int ld, int mb, int lane, int kstep) {
  int m  = mb + (lane & 15);
  int ko = kstep * 32 + (lane >> 4) * 8;
  AF f;
  f.q[0] = *(const uint4*)(base + m * ld + ko);
  f.q[1] = *(const uint4*)(base + m * ld + ko + 16);
  return f.v;
}
// B 32x16 f16 from weights stored transposed [N][Kpad] (K contiguous):
// lanes0-15 col N=lane, K: [0..7]+[16..23]; lanes16-31 col N=lane-16, K: [8..15]+[24..31]
__device__ __forceinline__ v16h load_b_glb(const _Float16* wt, int kpad, int j, int lane, int kstep) {
  int n  = j * 16 + (lane & 15);
  int kb = kstep * 32 + (lane >> 4) * 8;
  const _Float16* p = wt + (size_t)n * kpad + kb;
  AF f;
  f.q[0] = *(const uint4*)(p);
  f.q[1] = *(const uint4*)(p + 16);
  return f.v;
}

#define WMMA_ACC(a, b, c) \
  __builtin_amdgcn_wmma_f32_16x16x32_f16(false, (a), false, (b), (short)0, (c), false, false)

// silu via hardware rcp (v_rcp_f32) instead of IEEE divide chain
__device__ __forceinline__ float silu_f(float t) {
  return t * __builtin_amdgcn_rcpf(1.0f + __expf(-t));
}

__device__ __forceinline__ void bilerp8(const float* g, int HW, float u, float v, float* o) {
  float x = u * (float)(HW - 1);
  float y = v * (float)(HW - 1);
  int x0 = (int)floorf(x);
  int y0 = (int)floorf(y);
  x0 = x0 < 0 ? 0 : (x0 > HW - 2 ? HW - 2 : x0);
  y0 = y0 < 0 ? 0 : (y0 > HW - 2 ? HW - 2 : y0);
  float wx = x - (float)x0, wy = y - (float)y0;
  float w00 = (1.f - wx) * (1.f - wy), w01 = wx * (1.f - wy);
  float w10 = (1.f - wx) * wy,         w11 = wx * wy;
  const float* c00 = g + ((size_t)y0 * HW + x0) * 8;
  const float* c10 = c00 + (size_t)HW * 8;
#pragma unroll
  for (int d = 0; d < 8; ++d)
    o[d] = c00[d] * w00 + c00[8 + d] * w01 + c10[d] * w10 + c10[8 + d] * w11;
}

// Transpose+convert weights: dst[n*Kpad+k] = (k<Ksrc) ? src[k*Nn+n] : 0   (f32 -> f16)
__global__ void prep_wt(const float* __restrict__ src, _Float16* __restrict__ dst,
                        int Ksrc, int Nn, int Kpad) {
  int idx = blockIdx.x * blockDim.x + threadIdx.x;
  int total = Nn * Kpad;
  if (idx >= total) return;
  int n = idx / Kpad;
  int k = idx - n * Kpad;
  dst[idx] = (k < Ksrc) ? (_Float16)src[(size_t)k * Nn + n] : (_Float16)0.f;
}

__global__ __launch_bounds__(128)
void ngc_fused(const float* __restrict__ coords, const float* __restrict__ angles,
               const float* __restrict__ rho,    const float* __restrict__ rho_n,
               const float* __restrict__ curve_feats, const float* __restrict__ type_sample,
               const float* __restrict__ gt0, const float* __restrict__ gr0,
               const float* __restrict__ gt1, const float* __restrict__ gr1,
               const float* __restrict__ gt2, const float* __restrict__ gr2,
               const float* __restrict__ film0_fcb, const float* __restrict__ filmh_fcb,
               const float* __restrict__ gammab,    const float* __restrict__ betab,
               const float* __restrict__ decb1,     const float* __restrict__ decW2,
               const float* __restrict__ decb2,
               const _Float16* __restrict__ wG,  const _Float16* __restrict__ wB,
               const _Float16* __restrict__ wF0, const _Float16* __restrict__ wFH,
               const _Float16* __restrict__ wD1,
               float* __restrict__ out, int N) {
  __shared__ __align__(16) _Float16 s_xA[TILE * LDX];
  __shared__ __align__(16) _Float16 s_xB[TILE * LDX];
  __shared__ __align__(16) _Float16 s_feat[TILE * 64];
  __shared__ __align__(16) _Float16 s_cf[TILE * 64];
  float* s_h = (float*)s_xA;  // decoder hidden aliases dead x buffer (64*67*4 <= 64*136*2)

  const int t    = threadIdx.x;
  const int base = blockIdx.x * TILE;

  // ---------- Phase 0: grid featurization + curve_feats staging ----------
  {
    int pl = t & 63;
    int half = t >> 6;
    int p = base + pl;
    bool valid = p < N;
    _Float16* fr = s_feat + pl * 64;
    float cv = valid ? coords[p] : 0.f;
    float o[8];
    if (half == 0) {
      if (valid) {
        float uu = angles[p];
        fr[0] = (_Float16)rho[p];
        bilerp8(gt0, 128, uu, cv, o);
#pragma unroll
        for (int d = 0; d < 8; ++d) fr[1 + d] = (_Float16)o[d];
        bilerp8(gt1, 256, uu, cv, o);
#pragma unroll
        for (int d = 0; d < 8; ++d) fr[9 + d] = (_Float16)o[d];
        bilerp8(gt2, 512, uu, cv, o);
#pragma unroll
        for (int d = 0; d < 8; ++d) fr[17 + d] = (_Float16)o[d];
      } else {
#pragma unroll
        for (int k = 0; k < 25; ++k) fr[k] = (_Float16)0.f;
      }
#pragma unroll
      for (int k = 49; k < 64; ++k) fr[k] = (_Float16)0.f;  // pad K 49..63
    } else {
      if (valid) {
        float uu = rho_n[p];
        bilerp8(gr0, 128, uu, cv, o);
#pragma unroll
        for (int d = 0; d < 8; ++d) fr[25 + d] = (_Float16)o[d];
        bilerp8(gr1, 256, uu, cv, o);
#pragma unroll
        for (int d = 0; d < 8; ++d) fr[33 + d] = (_Float16)o[d];
        bilerp8(gr2, 512, uu, cv, o);
#pragma unroll
        for (int d = 0; d < 8; ++d) fr[41 + d] = (_Float16)o[d];
      } else {
#pragma unroll
        for (int k = 25; k < 49; ++k) fr[k] = (_Float16)0.f;
      }
    }
    // stage curve_feats -> f16 LDS (each thread: half a row); clamp row for tail
    int row = t >> 1, cb = (t & 1) * 32;
    int p2 = base + row;
    int p2c = p2 >= N ? N - 1 : p2;
    const float* src = curve_feats + (size_t)p2c * 64 + cb;
    _Float16* dst = s_cf + row * 64 + cb;
#pragma unroll
    for (int e = 0; e < 32; ++e) dst[e] = (_Float16)src[e];
  }
  __syncthreads();

  const int wave = t >> 5;
  const int lane = t & 31;
  const int mb = wave * 16;
  const int lr = lane & 15;   // D-layout: column within tile
  const int hg = lane >> 4;   // D-layout: row group (0: M=v, 1: M=v+8)

  // curve_feats A-fragments (K=64), reused by every gamma/beta GEMM
  v16h acf0 = load_a_lds(s_cf, 64, mb, lane, 0);
  v16h acf1 = load_a_lds(s_cf, 64, mb, lane, 1);

  // ---------- Layer 0: x = silu(gamma0*(feat@W0+b0)+beta0) + type_sample ----------
  {
    v16h af0 = load_a_lds(s_feat, 64, mb, lane, 0);
    v16h af1 = load_a_lds(s_feat, 64, mb, lane, 1);
    // clamped row for branch-free type_sample loads on the tail tile
    const int rbase = mb + 8 * hg;
#pragma unroll
    for (int j = 0; j < 8; ++j) {
      v8f ag = {}; v8f ab = {}; v8f az = {};
      ag = WMMA_ACC(acf0, load_b_glb(wG, 64, j, lane, 0), ag);
      ag = WMMA_ACC(acf1, load_b_glb(wG, 64, j, lane, 1), ag);
      ab = WMMA_ACC(acf0, load_b_glb(wB, 64, j, lane, 0), ab);
      ab = WMMA_ACC(acf1, load_b_glb(wB, 64, j, lane, 1), ab);
      az = WMMA_ACC(af0,  load_b_glb(wF0, 64, j, lane, 0), az);
      az = WMMA_ACC(af1,  load_b_glb(wF0, 64, j, lane, 1), az);
      int n = j * 16 + lr;
      float fb = film0_fcb[n];
      float gb = gammab[n];
      float bb = betab[n];
      CF cg, cb2, cz; cg.v = ag; cb2.v = ab; cz.v = az;
#pragma unroll
      for (int v = 0; v < 8; ++v) {
        int rl = rbase + v;
        int p  = base + rl;
        int pc = p >= N ? N - 1 : p;           // branch-free clamp
        float tt  = (cg.f[v] + gb) * (cz.f[v] + fb) + (cb2.f[v] + bb);
        float res = type_sample[(size_t)pc * 128 + n];
        s_xB[rl * LDX + n] = (_Float16)(silu_f(tt) + res);
      }
    }
  }
  __syncthreads();

  // ---------- Residual FiLM blocks ----------
  _Float16* cur = s_xB;
  _Float16* nxt = s_xA;
#pragma unroll 1
  for (int i = 0; i < 2; ++i) {
    const _Float16* wg = wG + (size_t)(i + 1) * 128 * 64;
    const _Float16* wb = wB + (size_t)(i + 1) * 128 * 64;
    const _Float16* wf = wFH + (size_t)i * 128 * 128;
    const float* gbp = gammab + (i + 1) * 128;
    const float* bbp = betab + (i + 1) * 128;
    const float* fbp = filmh_fcb + i * 128;
    v16h ax0 = load_a_lds(cur, LDX, mb, lane, 0);
    v16h ax1 = load_a_lds(cur, LDX, mb, lane, 1);
    v16h ax2 = load_a_lds(cur, LDX, mb, lane, 2);
    v16h ax3 = load_a_lds(cur, LDX, mb, lane, 3);
#pragma unroll
    for (int j = 0; j < 8; ++j) {
      v8f ag = {}; v8f ab2 = {}; v8f az = {};
      ag  = WMMA_ACC(acf0, load_b_glb(wg, 64, j, lane, 0), ag);
      ag  = WMMA_ACC(acf1, load_b_glb(wg, 64, j, lane, 1), ag);
      ab2 = WMMA_ACC(acf0, load_b_glb(wb, 64, j, lane, 0), ab2);
      ab2 = WMMA_ACC(acf1, load_b_glb(wb, 64, j, lane, 1), ab2);
      az  = WMMA_ACC(ax0,  load_b_glb(wf, 128, j, lane, 0), az);
      az  = WMMA_ACC(ax1,  load_b_glb(wf, 128, j, lane, 1), az);
      az  = WMMA_ACC(ax2,  load_b_glb(wf, 128, j, lane, 2), az);
      az  = WMMA_ACC(ax3,  load_b_glb(wf, 128, j, lane, 3), az);
      int n = j * 16 + lr;
      float fb = fbp[n], gb = gbp[n], bb = bbp[n];
      CF cg, cbu, cz; cg.v = ag; cbu.v = ab2; cz.v = az;
#pragma unroll
      for (int v = 0; v < 8; ++v) {
        int rl = mb + 8 * hg + v;
        float tt = (cg.f[v] + gb) * (cz.f[v] + fb) + (cbu.f[v] + bb);
        float xo = (float)cur[rl * LDX + n];
        nxt[rl * LDX + n] = (_Float16)(xo + silu_f(tt));
      }
    }
    __syncthreads();
    _Float16* tmp = cur; cur = nxt; nxt = tmp;
  }

  // ---------- Decoder: h = relu(x@decW1+b1) (WMMA), sdf = h@decW2+b2 ----------
  {
    v16h ax0 = load_a_lds(cur, LDX, mb, lane, 0);
    v16h ax1 = load_a_lds(cur, LDX, mb, lane, 1);
    v16h ax2 = load_a_lds(cur, LDX, mb, lane, 2);
    v16h ax3 = load_a_lds(cur, LDX, mb, lane, 3);
#pragma unroll
    for (int j = 0; j < 4; ++j) {
      v8f az = {};
      az = WMMA_ACC(ax0, load_b_glb(wD1, 128, j, lane, 0), az);
      az = WMMA_ACC(ax1, load_b_glb(wD1, 128, j, lane, 1), az);
      az = WMMA_ACC(ax2, load_b_glb(wD1, 128, j, lane, 2), az);
      az = WMMA_ACC(ax3, load_b_glb(wD1, 128, j, lane, 3), az);
      int n = j * 16 + lr;
      float b1 = decb1[n];
      CF cz; cz.v = az;
#pragma unroll
      for (int v = 0; v < 8; ++v) {
        int rl = mb + 8 * hg + v;
        s_h[rl * LDH + n] = fmaxf(cz.f[v] + b1, 0.f);
      }
    }
  }
  __syncthreads();
  if (t < TILE) {
    int p = base + t;
    if (p < N) {
      float sum = decb2[0];
#pragma unroll 8
      for (int i = 0; i < 64; ++i) sum += s_h[t * LDH + i] * decW2[i];
      out[p] = sum;
    }
  }
}

extern "C" void kernel_launch(void* const* d_in, const int* in_sizes, int n_in,
                              void* d_out, int out_size, void* d_ws, size_t ws_size,
                              hipStream_t stream) {
  // setup_inputs() dict order (gt/gr interleaved per level):
  const float* coords      = (const float*)d_in[0];
  const float* angles      = (const float*)d_in[1];
  const float* rho         = (const float*)d_in[2];
  const float* rho_n       = (const float*)d_in[3];
  const float* curve_feats = (const float*)d_in[4];
  const float* type_sample = (const float*)d_in[5];
  const float* gt0 = (const float*)d_in[6];
  const float* gr0 = (const float*)d_in[7];
  const float* gt1 = (const float*)d_in[8];
  const float* gr1 = (const float*)d_in[9];
  const float* gt2 = (const float*)d_in[10];
  const float* gr2 = (const float*)d_in[11];
  const float* film0_fcW = (const float*)d_in[12];
  const float* film0_fcb = (const float*)d_in[13];
  const float* filmh_fcW = (const float*)d_in[14];
  const float* filmh_fcb = (const float*)d_in[15];
  const float* gammaW    = (const float*)d_in[16];
  const float* gammab    = (const float*)d_in[17];
  const float* betaW     = (const float*)d_in[18];
  const float* betab     = (const float*)d_in[19];
  const float* decW1     = (const float*)d_in[20];
  const float* decb1     = (const float*)d_in[21];
  const float* decW2     = (const float*)d_in[22];
  const float* decb2     = (const float*)d_in[23];

  const int N = in_sizes[0];

  // f16 weight workspace layout (total 98304 f16 = 192 KB)
  _Float16* ws  = (_Float16*)d_ws;
  _Float16* wG  = ws;                    // gammaW^T  [3][128][64]
  _Float16* wB  = wG + 3 * 128 * 64;     // betaW^T   [3][128][64]
  _Float16* wF0 = wB + 3 * 128 * 64;     // film0^T   [128][64] (K padded 49->64)
  _Float16* wFH = wF0 + 128 * 64;        // filmh^T   [2][128][128]
  _Float16* wD1 = wFH + 2 * 128 * 128;   // decW1^T   [64][128]

  const int thr = 256;
  for (int l = 0; l < 3; ++l) {
    prep_wt<<<(128 * 64 + thr - 1) / thr, thr, 0, stream>>>(
        gammaW + (size_t)l * 64 * 128, wG + (size_t)l * 128 * 64, 64, 128, 64);
    prep_wt<<<(128 * 64 + thr - 1) / thr, thr, 0, stream>>>(
        betaW + (size_t)l * 64 * 128, wB + (size_t)l * 128 * 64, 64, 128, 64);
  }
  prep_wt<<<(128 * 64 + thr - 1) / thr, thr, 0, stream>>>(film0_fcW, wF0, 49, 128, 64);
  for (int i = 0; i < 2; ++i) {
    prep_wt<<<(128 * 128 + thr - 1) / thr, thr, 0, stream>>>(
        filmh_fcW + (size_t)i * 128 * 128, wFH + (size_t)i * 128 * 128, 128, 128, 128);
  }
  prep_wt<<<(64 * 128 + thr - 1) / thr, thr, 0, stream>>>(decW1, wD1, 128, 64, 128);

  int blocks = (N + TILE - 1) / TILE;
  ngc_fused<<<blocks, 128, 0, stream>>>(
      coords, angles, rho, rho_n, curve_feats, type_sample,
      gt0, gr0, gt1, gr1, gt2, gr2,
      film0_fcb, filmh_fcb, gammab, betab, decb1, decW2, decb2,
      wG, wB, wF0, wFH, wD1,
      (float*)d_out, N);
}
